// SpatialLSTM_85993835200892
// MI455X (gfx1250) — matrix operand order repaired and checked by reading, not compiled, not run on hardware
//
#include <hip/hip_runtime.h>
#include <hip/hip_bf16.h>
#include <stdint.h>

// ---------------------------------------------------------------------------
// SpatialLSTM on MI455X (gfx1250, wave32):
//   - gate GEMMs via v_wmma_f32_16x16x32_bf16 (fp32 accumulate)
//   - x pre-transposed to cell-major bf16 so each (cell,t) tile is contiguous
//   - per-wave persistent B-fragments (weights) held in VGPRs across T steps
//   - elementwise gate math on V_TANH_F32 (single TRANS op, co-executes with VALU)
//   - h stored bf16 cell-major (48 MB -> resident in 192 MB L2 for the conv)
// ---------------------------------------------------------------------------

typedef __attribute__((ext_vector_type(16))) __bf16 v16bf;
typedef __attribute__((ext_vector_type(8)))  __bf16 v8bf;
typedef __attribute__((ext_vector_type(8)))  float  v8f;

#define Bsz 32
#define Tt  24
#define Dd  32
#define Hh  32
#define Cc  1024   // M*N

__device__ __forceinline__ float tanh_hw(float x) {
#if __has_builtin(__builtin_amdgcn_tanhf)
    return __builtin_amdgcn_tanhf(x);                       // v_tanh_f32 (gfx1250 TRANS)
#else
    return 2.0f * __builtin_amdgcn_rcpf(1.0f + __expf(-2.0f * x)) - 1.0f;
#endif
}
__device__ __forceinline__ float sigf(float x) {            // sigmoid via tanh: 1 TRANS + 1 FMA
    return __builtin_fmaf(tanh_hw(0.5f * x), 0.5f, 0.5f);
}

// ---------------- kernel 1: fp32 -> bf16 weight convert ----------------
__global__ void __launch_bounds__(256)
cvt_f32_bf16(const float* __restrict__ src, unsigned short* __restrict__ dst, int n) {
    __bf16* d = reinterpret_cast<__bf16*>(dst);
    int base = blockIdx.x * 1024 + threadIdx.x;
#pragma unroll
    for (int j = 0; j < 4; ++j) {
        int k = base + j * 256;
        if (k < n) d[k] = (__bf16)src[k];
    }
}

// ---------------- kernel 2: x (B,T,D,cell) -> xT bf16 (cell,T,B,D) ----------------
__global__ void __launch_bounds__(256)
xpose(const float* __restrict__ x, unsigned short* __restrict__ xT_u) {
    __bf16* xT = reinterpret_cast<__bf16*>(xT_u);
    __shared__ float tile[32][33];                 // +1 pad: conflict-free transpose
    int bx = blockIdx.x;
    int ctile = bx & 31;                           // 32 cells per block
    int t = (bx >> 5) % Tt;
    int b = bx / (32 * Tt);
    int tid = threadIdx.x;

    // coalesced load: rows = d, cols = 32 consecutive cells
    int cc = tid & 31, q = tid >> 5;
    const float* src = x + ((size_t)(b * Tt + t) * Dd) * Cc + ctile * 32;
#pragma unroll
    for (int r = 0; r < 4; ++r) {
        int d = q + r * 8;
        tile[d][cc] = src[(size_t)d * Cc + cc];
    }
    __syncthreads();
    // coalesced store: contiguous over d per cell row
    int d = tid & 31, c0 = tid >> 5;
#pragma unroll
    for (int r = 0; r < 4; ++r) {
        int cl = c0 + r * 8;
        int cell = ctile * 32 + cl;
        xT[((size_t)(cell * Tt + t) * Bsz + b) * Dd + d] = (__bf16)tile[d][cl];
    }
}

// ---------------- kernel 3: per-cell LSTM over T steps (WMMA) ----------------
__global__ void __launch_bounds__(256)
lstm_cell(const unsigned short* __restrict__ xT_u,
          const unsigned short* __restrict__ Wih_u,
          const unsigned short* __restrict__ Whh_u,
          const float* __restrict__ b_ih,
          const float* __restrict__ b_hh,
          unsigned short* __restrict__ hS_u) {
    const __bf16* xT = reinterpret_cast<const __bf16*>(xT_u);
    __bf16* hS = reinterpret_cast<__bf16*>(hS_u);

    __shared__ __align__(16) __bf16 Wih_l[128 * 32];   // 8 KB
    __shared__ __align__(16) __bf16 Whh_l[128 * 32];   // 8 KB
    __shared__ __align__(16) __bf16 xt_l[1024];        // 2 KB  (32b x 32d)
    __shared__ __align__(16) __bf16 hb_l[1024];        // 2 KB  (recurrent A-matrix)
    __shared__ float gates_l[32 * 128];                // 16 KB
    __shared__ float c_l[1024];                        // 4 KB
    __shared__ float bias_l[128];

    const int cell = blockIdx.x;
    const int tid  = threadIdx.x;
    const int lane = tid & 31;        // wave32
    const int wid  = tid >> 5;        // 8 waves; wave = one 16-col gate tile
    const int rlo  = lane & 15;
    const int hsel = lane >> 4;       // lane-group: K-half select

    // stage bf16 weights for this cell (4096 halves each = 8 KB)
    {
        const uint4* gi = (const uint4*)(Wih_u + (size_t)cell * 4096);
        const uint4* gh = (const uint4*)(Whh_u + (size_t)cell * 4096);
        uint4* li = (uint4*)Wih_l;  uint4* lh = (uint4*)Whh_l;
        li[tid] = gi[tid];  li[tid + 256] = gi[tid + 256];
        lh[tid] = gh[tid];  lh[tid + 256] = gh[tid + 256];
    }
    if (tid < 128) bias_l[tid] = b_ih[cell * 128 + tid] + b_hh[cell * 128 + tid];
#pragma unroll
    for (int j = 0; j < 4; ++j) { c_l[tid + 256 * j] = 0.0f; hb_l[tid + 256 * j] = (__bf16)0.0f; }
    __syncthreads();

    // Persistent B fragments (same K order as A: lanes 0-15 = K{0..7,16..23}, lanes 16-31 = K{8..15,24..31})
    v16bf wihB, whhB;
    {
        int g = wid * 16 + rlo;          // gate column (row of W)
        int kb = hsel * 8;
        const v8bf* pi = (const v8bf*)(Wih_l + g * 32 + kb);
        v8bf a0 = pi[0], a1 = pi[2];
        wihB = __builtin_shufflevector(a0, a1, 0,1,2,3,4,5,6,7,8,9,10,11,12,13,14,15);
        const v8bf* ph = (const v8bf*)(Whh_l + g * 32 + kb);
        v8bf b0 = ph[0], b1 = ph[2];
        whhB = __builtin_shufflevector(b0, b1, 0,1,2,3,4,5,6,7,8,9,10,11,12,13,14,15);
    }
    const float bv = bias_l[wid * 16 + rlo];
    const v8f cini = {bv, bv, bv, bv, bv, bv, bv, bv};
    const int kbA = hsel * 8;            // A-matrix K base: lanes0-15 {0..7,16..23}, lanes16-31 {8..15,24..31}
    const __bf16* xbase = xT + (size_t)cell * Tt * 1024;

    for (int t = 0; t < Tt; ++t) {
        // stage x_t (contiguous 2 KB tile), 8 B per thread
        ((uint2*)xt_l)[tid] = ((const uint2*)(xbase + t * 1024))[tid];
        if (t + 1 < Tt) __builtin_prefetch(xbase + (t + 1) * 1024 + tid * 4, 0, 0);
        __syncthreads();   // S1: x_t staged, h_{t-1} (hb_l) final

#pragma unroll
        for (int mt = 0; mt < 2; ++mt) {
            int row = mt * 16 + rlo;
            // A fragments per ISA 16-bit A layout: two contiguous 16B chunks per lane
            const __bf16* xr = xt_l + row * 32 + kbA;
            v8bf x0 = *(const v8bf*)xr;
            v8bf x1 = *(const v8bf*)(xr + 16);
            v16bf xA = __builtin_shufflevector(x0, x1, 0,1,2,3,4,5,6,7,8,9,10,11,12,13,14,15);
            const __bf16* hr = hb_l + row * 32 + kbA;
            v8bf h0 = *(const v8bf*)hr;
            v8bf h1 = *(const v8bf*)(hr + 16);
            v16bf hA = __builtin_shufflevector(h0, h1, 0,1,2,3,4,5,6,7,8,9,10,11,12,13,14,15);

            v8f acc = cini;
            acc = __builtin_amdgcn_wmma_f32_16x16x32_bf16(false, hA, false, whhB, (short)0, acc, false, false);
            acc = __builtin_amdgcn_wmma_f32_16x16x32_bf16(false, xA, false, wihB, (short)0, acc, false, false);

            // D layout: VGPR r -> row r (+8 for lanes 16-31), col = lane&15
            int col = wid * 16 + rlo;
            int r0  = mt * 16 + hsel * 8;
#pragma unroll
            for (int r = 0; r < 8; ++r) gates_l[(r0 + r) * 128 + col] = acc[r];
        }
        __syncthreads();   // S2: all gates in LDS

        // fused elementwise LSTM cell update; thread owns fixed (b,h) elements
        size_t hout = (size_t)(cell * Tt + t) * 1024;
#pragma unroll
        for (int j = 0; j < 4; ++j) {
            int e  = tid + 256 * j;
            int bb = e >> 5, hh = e & 31;
            float gi = gates_l[bb * 128 + hh];
            float gf = gates_l[bb * 128 + 32 + hh];
            float gg = gates_l[bb * 128 + 64 + hh];
            float go = gates_l[bb * 128 + 96 + hh];
            float cn = sigf(gf) * c_l[e] + sigf(gi) * tanh_hw(gg);
            c_l[e] = cn;
            float hf = sigf(go) * tanh_hw(cn);
            hb_l[e] = (__bf16)hf;                 // recurrent input for t+1
            hS[hout + e] = (__bf16)hf;            // cell-major (C,T,B,H), coalesced
        }
        // no barrier needed: next iter only writes xt_l (disjoint) before S1
    }
}

// ---------------- kernel 4: 3x3xH conv + sigmoid head ----------------
__global__ void __launch_bounds__(256)
conv_head(const unsigned short* __restrict__ hS_u,
          const float* __restrict__ cw,
          const float* __restrict__ cb,
          float* __restrict__ out) {
    const __bf16* hS = reinterpret_cast<const __bf16*>(hS_u);
    __shared__ float w_l[288];
    int tid = threadIdx.x;
    for (int j = tid; j < 288; j += 256) w_l[j] = cw[j];
    __syncthreads();

    int idx = blockIdx.x * 256 + tid;          // flat (B,T,M,N)
    int r = idx & 1023;
    int bt = idx >> 10;
    int n = r & 31, m = r >> 5;
    int t = bt % Tt, b = bt / Tt;

    float acc = cb[0];
#pragma unroll
    for (int dm = 0; dm < 3; ++dm) {
        int mm = m + dm - 1;
        if (mm < 0 || mm >= 32) continue;
#pragma unroll
        for (int dn = 0; dn < 3; ++dn) {
            int nn = n + dn - 1;
            if (nn < 0 || nn >= 32) continue;
            const v8bf* pv = (const v8bf*)(hS + ((size_t)((mm * 32 + nn) * Tt + t) * Bsz + b) * Hh);
            float s = 0.0f;
#pragma unroll
            for (int c = 0; c < 4; ++c) {
                v8bf v = pv[c];
#pragma unroll
                for (int j = 0; j < 8; ++j)
                    s += (float)v[j] * w_l[(c * 8 + j) * 9 + dm * 3 + dn];
            }
            acc += s;
        }
    }
    out[idx] = sigf(acc);
}

// ---------------- launch ----------------
extern "C" void kernel_launch(void* const* d_in, const int* in_sizes, int n_in,
                              void* d_out, int out_size, void* d_ws, size_t ws_size,
                              hipStream_t stream) {
    const float* x      = (const float*)d_in[0];   // (B,T,D,M,N)
    const float* W_ih   = (const float*)d_in[1];   // (C,4H,D)
    const float* W_hh   = (const float*)d_in[2];   // (C,4H,H)
    const float* b_ih_p = (const float*)d_in[3];   // (C,4H)
    const float* b_hh_p = (const float*)d_in[4];   // (C,4H)
    const float* conv_w = (const float*)d_in[5];   // (1,H,3,3)
    const float* conv_b = (const float*)d_in[6];   // (1,)
    float* out = (float*)d_out;

    // workspace layout (118 MB total)
    char* ws = (char*)d_ws;
    unsigned short* xT   = (unsigned short*)(ws);             // 50,331,648 B: bf16 (C,T,B,D)
    unsigned short* Wihb = (unsigned short*)(ws + 50331648);  //  8,388,608 B
    unsigned short* Whhb = (unsigned short*)(ws + 58720256);  //  8,388,608 B
    unsigned short* hS   = (unsigned short*)(ws + 67108864);  // 50,331,648 B: bf16 (C,T,B,H)

    const int nW = Cc * 128 * 32;                  // 4,194,304 per weight tensor
    cvt_f32_bf16<<<(nW + 1023) / 1024, 256, 0, stream>>>(W_ih, Wihb, nW);
    cvt_f32_bf16<<<(nW + 1023) / 1024, 256, 0, stream>>>(W_hh, Whhb, nW);
    xpose<<<Bsz * Tt * 32, 256, 0, stream>>>(x, xT);
    lstm_cell<<<Cc, 256, 0, stream>>>(xT, Wihb, Whhb, b_ih_p, b_hh_p, hS);
    conv_head<<<(Bsz * Tt * Cc) / 256, 256, 0, stream>>>(hS, conv_w, conv_b, out);
}
